// CdwExtractor_18665927868989
// MI455X (gfx1250) — matrix operand order repaired and checked by reading, not compiled, not run on hardware
//
#include <hip/hip_runtime.h>
#include <hip/hip_bf16.h>

#define BATCH 2
#define NPTS  8192
#define KNN_K 16
#define MROWS (BATCH * NPTS)

typedef __attribute__((ext_vector_type(16))) _Float16     v16h;
typedef __attribute__((ext_vector_type(8)))  float        v8f;
typedef __attribute__((ext_vector_type(4)))  unsigned int u32x4;
typedef __attribute__((ext_vector_type(8)))  int          i32x8;
typedef __attribute__((ext_vector_type(4)))  int          i32x4;

#if defined(__has_builtin)
#if __has_builtin(__builtin_amdgcn_tensor_load_to_lds)
#define HAVE_TDM 1
#endif
#endif

// ---------------------------------------------------------------------------
// Wave-tiled WMMA GEMM: C[M,Nd] = act( A[M,Kd] @ W[Kd,Nd] + bias (+R) )
// Block = 256 threads (8 waves). blockIdx.y picks the 16-col N tile; the
// block's W column panel (Kd x 16 f32, up to 64KB) is staged into LDS by the
// Tensor Data Mover (D# per CDNA5 ISA), then each wave
// computes FOUR 16x16 C tiles (512 rows/block), reusing one B fragment per
// K-step across 4 v_wmma_f32_16x16x32_f16. Kd must be a multiple of 32.
// Row-masked so the M=2 FC head works on the same kernel.
// ---------------------------------------------------------------------------
__global__ void gemm_wmma_f16(const float* __restrict__ A,
                              const float* __restrict__ W,
                              const float* __restrict__ bias,
                              const float* __restrict__ R,   // optional residual
                              float* __restrict__ C,
                              int M, int Kd, int Nd,
                              int relu_flag, int has_res)
{
    __shared__ float sW[16 * 1024];          // Kd(<=1024) x 16 panel, row-major 16 wide

    const int tn = blockIdx.y;               // N tile (16 columns)

#ifdef HAVE_TDM
    if (threadIdx.x == 0) {
        // --- Tensor DMA descriptor: 2-D tile, Kd rows x 16 cols of f32 ---
        unsigned long long ga  = (unsigned long long)(const void*)(W + (size_t)tn * 16);
        unsigned int lds_off   = (unsigned int)(unsigned long long)(const void*)&sW[0];
        u32x4 g0;
        g0[0] = 1u;                                       // count=1, user descriptor
        g0[1] = lds_off;                                  // lds_addr (bytes)
        g0[2] = (unsigned int)ga;                         // global_addr[31:0]
        g0[3] = (unsigned int)((ga >> 32) & 0x1FFFFFFu)   // global_addr[56:32]
                | (2u << 30);                             // type=2 ("image")
        i32x8 g1;
        g1[0] = 0x00020000;                               // workgroup_mask=0, data_size=4B
        g1[1] = (int)(((unsigned)Nd & 0xFFFFu) << 16);    // tensor_dim0[15:0] @ bits 63:48
        g1[2] = (int)((((unsigned)Nd >> 16) & 0xFFFFu)    // tensor_dim0[31:16]
                | (((unsigned)Kd & 0xFFFFu) << 16));      // tensor_dim1[15:0]
        g1[3] = (int)((((unsigned)Kd >> 16) & 0xFFFFu)    // tensor_dim1[31:16]
                | (16u << 16));                           // tile_dim0 = 16
        g1[4] = (int)((unsigned)Kd & 0xFFFFu);            // tile_dim1 = Kd, tile_dim2 = 0
        g1[5] = (int)(unsigned)Nd;                        // tensor_dim0_stride[31:0] = Nd
        g1[6] = 0;                                        // stride hi, dim1_stride lo
        g1[7] = 0;
        i32x4 gz  = {0, 0, 0, 0};                         // groups 2/3 unused (2-D tensor)
        i32x8 gz8 = {0, 0, 0, 0, 0, 0, 0, 0};             // extra group (clang-23 form)
        __builtin_amdgcn_tensor_load_to_lds(g0, g1, gz, gz, gz8, 0);
        __builtin_amdgcn_s_wait_tensorcnt(0);
    }
    __syncthreads();
#else
    for (int i = threadIdx.x; i < Kd * 16; i += blockDim.x)
        sW[i] = W[(size_t)(i >> 4) * Nd + (size_t)tn * 16 + (i & 15)];
    __syncthreads();
#endif

    const int wave = threadIdx.x >> 5;
    const int lane = threadIdx.x & 31;
    const int halfsel = (lane >> 4) & 1;     // 0: lanes 0-15, 1: lanes 16-31
    const int l15  = lane & 15;
    const int koffA = halfsel * 8;           // A: low half K 0..7/16..23, high 8..15/24..31
    const int bcol  = tn * 16 + l15;
    const int rowBase = (blockIdx.x * 8 + wave) * 64;   // 4 M-tiles per wave

    v8f acc0 = {}, acc1 = {}, acc2 = {}, acc3 = {};

    for (int kk = 0; kk < Kd; kk += 32) {
        // B fragment from LDS panel (32x16): lanes 0-15 K=kk+0..7/16..23, 16-31 K=kk+8..15/24..31
        const float* bp = &sW[(kk + halfsel * 8) * 16 + l15];
        v16h b;
        #pragma unroll
        for (int j = 0; j < 16; ++j) b[j] = (_Float16)bp[(j + (j >= 8 ? 8 : 0)) * 16];

        v8f* accs[4] = { &acc0, &acc1, &acc2, &acc3 };
        #pragma unroll
        for (int t = 0; t < 4; ++t) {
            const int r0 = rowBase + t * 16 + l15;
            v16h a;
            if (r0 < M) {
                const float* ap = A + (size_t)r0 * Kd + kk + koffA;
                #pragma unroll
                for (int j = 0; j < 8; ++j) a[j]     = (_Float16)ap[j];
                #pragma unroll
                for (int j = 0; j < 8; ++j) a[8 + j] = (_Float16)ap[16 + j];
            } else {
                #pragma unroll
                for (int j = 0; j < 16; ++j) a[j] = (_Float16)0.0f;
            }
            *accs[t] = __builtin_amdgcn_wmma_f32_16x16x32_f16(
                           false, a, false, b, (short)0, *accs[t], false, false);
        }
    }

    const float bb = bias[bcol];
    v8f* accs[4] = { &acc0, &acc1, &acc2, &acc3 };
    #pragma unroll
    for (int t = 0; t < 4; ++t) {
        #pragma unroll
        for (int r = 0; r < 8; ++r) {
            const int m = rowBase + t * 16 + halfsel * 8 + r;  // C: VGPR r = rows r / r+8
            if (m < M) {
                float v = (*accs[t])[r] + bb;
                if (has_res)  v += R[(size_t)m * Nd + bcol];
                if (relu_flag) v = fmaxf(v, 0.0f);
                C[(size_t)m * Nd + bcol] = v;
            }
        }
    }
}

// ---------------------------------------------------------------------------
// KNN: one thread per query point; LDS-tiled candidates; register
// insertion-sort top-(K+1) by squared distance (self included, dist 0).
// ---------------------------------------------------------------------------
__global__ void knn_kernel(const float* __restrict__ pts, int* __restrict__ knn)
{
    const int b   = blockIdx.y;
    const int n   = blockIdx.x * blockDim.x + threadIdx.x;
    const int tid = threadIdx.x;
    __shared__ float sp[128 * 3];

    const float* P = pts + (size_t)b * NPTS * 3;
    const float px = P[n * 3 + 0], py = P[n * 3 + 1], pz = P[n * 3 + 2];

    float bd[KNN_K + 1];
    int   bi[KNN_K + 1];
    #pragma unroll
    for (int t = 0; t <= KNN_K; ++t) { bd[t] = 3.4e38f; bi[t] = 0; }

    for (int t0 = 0; t0 < NPTS; t0 += 128) {
        __syncthreads();
        sp[tid * 3 + 0] = P[(t0 + tid) * 3 + 0];
        sp[tid * 3 + 1] = P[(t0 + tid) * 3 + 1];
        sp[tid * 3 + 2] = P[(t0 + tid) * 3 + 2];
        __syncthreads();
        for (int jj = 0; jj < 128; ++jj) {
            const float dx = px - sp[jj * 3 + 0];
            const float dy = py - sp[jj * 3 + 1];
            const float dz = pz - sp[jj * 3 + 2];
            const float d  = dx * dx + dy * dy + dz * dz;
            if (d < bd[KNN_K]) {
                bd[KNN_K] = d; bi[KNN_K] = t0 + jj;
                #pragma unroll
                for (int t = KNN_K; t > 0; --t) {
                    if (bd[t] < bd[t - 1]) {
                        float td = bd[t]; bd[t] = bd[t - 1]; bd[t - 1] = td;
                        int   ti = bi[t]; bi[t] = bi[t - 1]; bi[t - 1] = ti;
                    }
                }
            }
        }
    }
    int* o = knn + ((size_t)b * NPTS + n) * (KNN_K + 1);
    #pragma unroll
    for (int t = 0; t <= KNN_K; ++t) o[t] = bi[t];
}

// ---------------------------------------------------------------------------
// Neighbor MLP + pooling + ftr_1. One thread per point; weights in LDS.
// ---------------------------------------------------------------------------
__global__ void ftr1_kernel(const float* __restrict__ pts,
                            const int* __restrict__ knn,
                            const float* __restrict__ w1, const float* __restrict__ b1,
                            const float* __restrict__ w2, const float* __restrict__ b2,
                            const float* __restrict__ w2n, const float* __restrict__ b2n,
                            const float* __restrict__ w3, const float* __restrict__ b3,
                            float* __restrict__ ftr1)
{
    __shared__ float s_w1[7 * 16], s_b1[16];
    __shared__ float s_w2[16 * 32], s_b2[32];
    __shared__ float s_wn[3 * 32], s_bn[32];
    __shared__ float s_w3[64 * 32], s_b3[32];
    const int tid = threadIdx.x;
    for (int i = tid; i < 7 * 16;  i += blockDim.x) s_w1[i] = w1[i];
    for (int i = tid; i < 16;      i += blockDim.x) s_b1[i] = b1[i];
    for (int i = tid; i < 16 * 32; i += blockDim.x) s_w2[i] = w2[i];
    for (int i = tid; i < 32;      i += blockDim.x) s_b2[i] = b2[i];
    for (int i = tid; i < 3 * 32;  i += blockDim.x) s_wn[i] = w2n[i];
    for (int i = tid; i < 32;      i += blockDim.x) s_bn[i] = b2n[i];
    for (int i = tid; i < 64 * 32; i += blockDim.x) s_w3[i] = w3[i];
    for (int i = tid; i < 32;      i += blockDim.x) s_b3[i] = b3[i];
    __syncthreads();

    const int b = blockIdx.y;
    const int n = blockIdx.x * blockDim.x + tid;
    const float* P = pts + (size_t)b * NPTS * 3;
    const int* idx = knn + ((size_t)b * NPTS + n) * (KNN_K + 1);

    const int i0 = idx[0];
    const float ax = P[i0 * 3 + 0], ay = P[i0 * 3 + 1], az = P[i0 * 3 + 2];

    float pooled[32];
    #pragma unroll
    for (int o = 0; o < 32; ++o) pooled[o] = -3.4e38f;

    for (int k = 1; k <= KNN_K; ++k) {
        const int j = idx[k];
        const float rx = P[j * 3 + 0] - ax;
        const float ry = P[j * 3 + 1] - ay;
        const float rz = P[j * 3 + 2] - az;
        const float dd = sqrtf(rx * rx + ry * ry + rz * rz + 1e-8f);
        float f[7] = { ax, ay, az, rx, ry, rz, dd };
        float h1[16];
        #pragma unroll
        for (int o = 0; o < 16; ++o) {
            float s = s_b1[o];
            #pragma unroll
            for (int i = 0; i < 7; ++i) s += f[i] * s_w1[i * 16 + o];
            h1[o] = fmaxf(s, 0.0f);
        }
        #pragma unroll
        for (int o = 0; o < 32; ++o) {
            float s = s_b2[o];
            #pragma unroll
            for (int i = 0; i < 16; ++i) s += h1[i] * s_w2[i * 32 + o];
            s = fmaxf(s, 0.0f);
            pooled[o] = fmaxf(pooled[o], s);
        }
    }

    const float px = P[n * 3 + 0], py = P[n * 3 + 1], pz = P[n * 3 + 2];
    float cat[64];
    #pragma unroll
    for (int o = 0; o < 32; ++o) {
        float s = s_bn[o] + px * s_wn[0 * 32 + o] + py * s_wn[1 * 32 + o] + pz * s_wn[2 * 32 + o];
        cat[o] = fmaxf(s, 0.0f);
        cat[32 + o] = pooled[o];
    }
    float* out = ftr1 + ((size_t)b * NPTS + n) * 32;
    #pragma unroll
    for (int o = 0; o < 32; ++o) {
        float s = s_b3[o];
        #pragma unroll
        for (int i = 0; i < 64; ++i) s += cat[i] * s_w3[i * 32 + o];
        out[o] = fmaxf(s, 0.0f);
    }
}

// ---------------------------------------------------------------------------
// Max over N (axis=1): one block per (b, c), strided loop + LDS tree reduce.
// ---------------------------------------------------------------------------
__global__ void reduce_max_kernel(const float* __restrict__ X, float* __restrict__ out, int C)
{
    const int b = blockIdx.x / C;
    const int c = blockIdx.x % C;
    const float* x = X + (size_t)b * NPTS * C + c;
    float m = -3.4e38f;
    for (int n = threadIdx.x; n < NPTS; n += blockDim.x)
        m = fmaxf(m, x[(size_t)n * C]);
    __shared__ float s[256];
    s[threadIdx.x] = m;
    __syncthreads();
    for (int off = 128; off > 0; off >>= 1) {
        if (threadIdx.x < off) s[threadIdx.x] = fmaxf(s[threadIdx.x], s[threadIdx.x + off]);
        __syncthreads();
    }
    if (threadIdx.x == 0) out[blockIdx.x] = s[0];
}

// ---------------------------------------------------------------------------
// Attention: per (b,c) sum(exp(l-lmax)) and sum(f4*exp(l-lmax)); att = ratio.
// ---------------------------------------------------------------------------
__global__ void att_reduce_kernel(const float* __restrict__ logits,
                                  const float* __restrict__ f4,
                                  const float* __restrict__ lmax,
                                  float* __restrict__ att, int C)
{
    const int b = blockIdx.x / C;
    const int c = blockIdx.x % C;
    const float lm = lmax[blockIdx.x];
    const float* l = logits + (size_t)b * NPTS * C + c;
    const float* f = f4     + (size_t)b * NPTS * C + c;
    float s1 = 0.0f, s2 = 0.0f;
    for (int n = threadIdx.x; n < NPTS; n += blockDim.x) {
        const float e = __expf(l[(size_t)n * C] - lm);
        s1 += e;
        s2 += f[(size_t)n * C] * e;
    }
    __shared__ float sh1[256], sh2[256];
    sh1[threadIdx.x] = s1; sh2[threadIdx.x] = s2;
    __syncthreads();
    for (int off = 128; off > 0; off >>= 1) {
        if (threadIdx.x < off) {
            sh1[threadIdx.x] += sh1[threadIdx.x + off];
            sh2[threadIdx.x] += sh2[threadIdx.x + off];
        }
        __syncthreads();
    }
    if (threadIdx.x == 0) att[blockIdx.x] = sh2[0] / sh1[0];
}

// ---------------------------------------------------------------------------
// Concats & feat assembly
// ---------------------------------------------------------------------------
__global__ void concat2_kernel(const float* __restrict__ f2, const float* __restrict__ g2,
                               float* __restrict__ out)
{
    const int i = blockIdx.x * blockDim.x + threadIdx.x;
    if (i >= MROWS * 128) return;
    const int m = i >> 7, c = i & 127;
    const int b = m / NPTS;
    out[i] = (c < 64) ? f2[(size_t)m * 64 + c] : g2[b * 64 + (c - 64)];
}

__global__ void concat4_kernel(const float* __restrict__ f1, const float* __restrict__ f2,
                               const float* __restrict__ f3, const float* __restrict__ g3,
                               float* __restrict__ out)
{
    const int i = blockIdx.x * blockDim.x + threadIdx.x;
    if (i >= MROWS * 352) return;
    const int m = i / 352, c = i % 352;
    const int b = m / NPTS;
    float v;
    if (c < 32)        v = f1[(size_t)m * 32 + c];
    else if (c < 96)   v = f2[(size_t)m * 64 + (c - 32)];
    else if (c < 224)  v = f3[(size_t)m * 128 + (c - 96)];
    else               v = g3[b * 128 + (c - 224)];
    out[i] = v;
}

__global__ void feat_kernel(const float* __restrict__ fmx, const float* __restrict__ att,
                            float* __restrict__ feat)
{
    const int i = blockIdx.x * blockDim.x + threadIdx.x;
    if (i >= BATCH * 1024) return;
    const int b = i / 1024, c = i % 1024;
    feat[i] = (c < 512) ? fmx[b * 512 + c] : att[b * 512 + (c - 512)];
}

// ---------------------------------------------------------------------------
// Host-side orchestration
// ---------------------------------------------------------------------------
static inline void launch_gemm(const float* A, const float* W, const float* bias,
                               const float* R, float* C, int M, int Kd, int Nd,
                               int relu_flag, int has_res, hipStream_t stream)
{
    const int tilesM = (M + 15) / 16;
    dim3 grid((tilesM + 31) / 32, Nd / 16);   // 8 waves x 4 M-tiles = 32 tiles / block
    gemm_wmma_f16<<<grid, 256, 0, stream>>>(A, W, bias, R, C, M, Kd, Nd, relu_flag, has_res);
}

extern "C" void kernel_launch(void* const* d_in, const int* in_sizes, int n_in,
                              void* d_out, int out_size, void* d_ws, size_t ws_size,
                              hipStream_t stream)
{
    const float* pts    = (const float*)d_in[0];
    const float* na1_w1 = (const float*)d_in[1];  const float* na1_b1 = (const float*)d_in[2];
    const float* na1_w2 = (const float*)d_in[3];  const float* na1_b2 = (const float*)d_in[4];
    const float* na2_w  = (const float*)d_in[5];  const float* na2_b  = (const float*)d_in[6];
    const float* na3_w  = (const float*)d_in[7];  const float* na3_b  = (const float*)d_in[8];
    const float* r1a_w  = (const float*)d_in[9];  const float* r1a_b  = (const float*)d_in[10];
    const float* r1b_w  = (const float*)d_in[11]; const float* r1b_b  = (const float*)d_in[12];
    const float* r1s_w  = (const float*)d_in[13]; const float* r1s_b  = (const float*)d_in[14];
    const float* r2a_w  = (const float*)d_in[15]; const float* r2a_b  = (const float*)d_in[16];
    const float* r2b_w  = (const float*)d_in[17]; const float* r2b_b  = (const float*)d_in[18];
    const float* r2s_w  = (const float*)d_in[19]; const float* r2s_b  = (const float*)d_in[20];
    const float* fuse_w = (const float*)d_in[21]; const float* fuse_b = (const float*)d_in[22];
    const float* att_w  = (const float*)d_in[23]; const float* att_b  = (const float*)d_in[24];
    const float* fc1_w  = (const float*)d_in[25]; const float* fc1_b  = (const float*)d_in[26];
    const float* fc2_w  = (const float*)d_in[27]; const float* fc2_b  = (const float*)d_in[28];
    const float* fc3_w  = (const float*)d_in[29]; const float* fc3_b  = (const float*)d_in[30];
    float* out = (float*)d_out;

    char* w = (char*)d_ws;
    auto carve = [&](size_t bytes) -> char* {
        char* p = w;
        w += (bytes + 255) & ~(size_t)255;
        return p;
    };
    const size_t M = MROWS;
    int*   knn    = (int*)  carve((size_t)M * (KNN_K + 1) * sizeof(int));
    float* ftr1   = (float*)carve(M * 32  * 4);
    float* t1     = (float*)carve(M * 64  * 4);
    float* ts     = (float*)carve(M * 64  * 4);
    float* ftr2   = (float*)carve(M * 64  * 4);
    float* g2     = (float*)carve(BATCH * 64 * 4);
    float* cat2   = (float*)carve(M * 128 * 4);
    float* t2     = (float*)carve(M * 128 * 4);
    float* t2s    = (float*)carve(M * 128 * 4);
    float* ftr3   = (float*)carve(M * 128 * 4);
    float* g3     = (float*)carve(BATCH * 128 * 4);
    float* cat4   = (float*)carve(M * 352 * 4);
    float* ftr4   = (float*)carve(M * 512 * 4);
    float* logits = (float*)carve(M * 512 * 4);
    float* lmax   = (float*)carve(BATCH * 512 * 4);
    float* fmx    = (float*)carve(BATCH * 512 * 4);
    float* attv   = (float*)carve(BATCH * 512 * 4);
    float* feat   = (float*)carve(BATCH * 1024 * 4);
    float* h1     = (float*)carve(BATCH * 512 * 4);
    float* h2     = (float*)carve(BATCH * 1024 * 4);
    (void)ws_size; (void)in_sizes; (void)n_in; (void)out_size;

    // 1) KNN (top-17 incl. self)
    {
        dim3 g(NPTS / 128, BATCH);
        knn_kernel<<<g, 128, 0, stream>>>(pts, knn);
    }
    // 2) neighbor MLP + pooling + ftr_1
    {
        dim3 g(NPTS / 128, BATCH);
        ftr1_kernel<<<g, 128, 0, stream>>>(pts, knn, na1_w1, na1_b1, na1_w2, na1_b2,
                                           na2_w, na2_b, na3_w, na3_b, ftr1);
    }
    // 3) res_smlp 1: 32 -> 64
    launch_gemm(ftr1, r1a_w, r1a_b, nullptr, t1,   M, 32, 64, 1, 0, stream);
    launch_gemm(ftr1, r1s_w, r1s_b, nullptr, ts,   M, 32, 64, 0, 0, stream);
    launch_gemm(t1,   r1b_w, r1b_b, ts,      ftr2, M, 64, 64, 1, 1, stream);
    // 4) g2 = max_n ftr2
    reduce_max_kernel<<<BATCH * 64, 256, 0, stream>>>(ftr2, g2, 64);
    // 5) cat2 = [ftr2 | g2]
    concat2_kernel<<<(MROWS * 128 + 255) / 256, 256, 0, stream>>>(ftr2, g2, cat2);
    // 6) res_smlp 2: 128 -> 128
    launch_gemm(cat2, r2a_w, r2a_b, nullptr, t2,   M, 128, 128, 1, 0, stream);
    launch_gemm(cat2, r2s_w, r2s_b, nullptr, t2s,  M, 128, 128, 0, 0, stream);
    launch_gemm(t2,   r2b_w, r2b_b, t2s,     ftr3, M, 128, 128, 1, 1, stream);
    // 7) g3 = max_n ftr3
    reduce_max_kernel<<<BATCH * 128, 256, 0, stream>>>(ftr3, g3, 128);
    // 8) cat4 = [ftr1 | ftr2 | ftr3 | g3]  (352 ch)
    concat4_kernel<<<(MROWS * 352 + 255) / 256, 256, 0, stream>>>(ftr1, ftr2, ftr3, g3, cat4);
    // 9) ftr4 = relu(cat4 @ fuse)  (352 -> 512)
    launch_gemm(cat4, fuse_w, fuse_b, nullptr, ftr4, M, 352, 512, 1, 0, stream);
    // 10) logits = ftr4 @ att  (512 -> 512)
    launch_gemm(ftr4, att_w, att_b, nullptr, logits, M, 512, 512, 0, 0, stream);
    // 11) softmax(axis=1) stats + attention pooling + channel max
    reduce_max_kernel<<<BATCH * 512, 256, 0, stream>>>(logits, lmax, 512);
    reduce_max_kernel<<<BATCH * 512, 256, 0, stream>>>(ftr4, fmx, 512);
    att_reduce_kernel<<<BATCH * 512, 256, 0, stream>>>(logits, ftr4, lmax, attv, 512);
    // 12) feat = [max | att]  (B, 1024)
    feat_kernel<<<(BATCH * 1024 + 255) / 256, 256, 0, stream>>>(fmx, attv, feat);
    // 13) FC head (M=2, row-masked WMMA tiles)
    launch_gemm(feat, fc1_w, fc1_b, nullptr, h1,  BATCH, 1024, 512,  1, 0, stream);
    launch_gemm(h1,   fc2_w, fc2_b, nullptr, h2,  BATCH, 512,  1024, 1, 0, stream);
    launch_gemm(h2,   fc3_w, fc3_b, nullptr, out, BATCH, 1024, 1024, 0, 0, stream);
}
